// RNN_16947940950406
// MI455X (gfx1250) — compile-verified, matrix operand order checked by reading, not run on hardware
//
#include <hip/hip_runtime.h>
#include <stddef.h>

// ---------------------------------------------------------------------------
// 2-layer tanh RNN on gfx1250 (MI455X), bf16 WMMA + TDM (tensor_load_to_lds).
//
// Shapes: B=64, S=512, H=1024, IN=256, L=2.
//   * Weights packed to bf16 WMMA B-fragment order once per call.
//   * Hidden state row-major bf16 [64,1024], double buffered.
//   * Per GEMM kernel: TDM DMAs the whole A slab (128KB) into LDS, waves read
//     A fragments via ds_load_b128; B fragments stream from L2.
//   * Each wave computes a 2x2 block of 16x16 tiles (4 independent fp32
//     accumulator chains) -> 2 loads per v_wmma_f32_16x16x32_bf16.
//   * Block = 8 waves = 64x128 output slab. Layer GEMMs: 8 blocks; head: 2.
// ---------------------------------------------------------------------------

typedef __attribute__((ext_vector_type(16))) __bf16 v16bf;
typedef __attribute__((ext_vector_type(8)))  __bf16 v8bf;
typedef __attribute__((ext_vector_type(8)))  float  v8f;
typedef __attribute__((ext_vector_type(4)))  unsigned int u32x4;
typedef __attribute__((ext_vector_type(8)))  int i32x8;
typedef __attribute__((ext_vector_type(4)))  int i32x4;

#define RNN_B   64
#define RNN_S   512
#define RNN_H   1024
#define RNN_IN  256
#define LDS_A_BYTES (RNN_B * RNN_H * 2)   // 131072

// ---------------------------------------------------------------------------
// Weight packing: row-major fp32 W[K][N] -> bf16 in WMMA-B fragment order.
// GEMM fetch for N-tile ntg, K-chunk kc: contiguous 1KB at
//   Wp + (ntg*(K/32) + kc)*512, lane reads 16 bf16 at +lane*16.
// ---------------------------------------------------------------------------
__global__ __launch_bounds__(256) void k_pack_b(const float* __restrict__ W,
                                                __bf16* __restrict__ out,
                                                int K, int N) {
    int gid = blockIdx.x * 256 + threadIdx.x;
    int total = K * N;
    if (gid >= total) return;
    int e    = gid & 15;
    int lane = (gid >> 4) & 31;
    int tile = gid >> 9;
    int kcN  = K >> 5;
    int kc   = tile % kcN;
    int nt   = tile / kcN;
    int n    = nt * 16 + (lane & 15);
    int kk   = (lane < 16) ? ((e < 8) ? e : e + 8)
                           : ((e < 8) ? e + 8 : e + 16);
    int k    = kc * 32 + kk;
    out[gid] = (__bf16)W[(size_t)k * N + n];
}

__global__ __launch_bounds__(256) void k_f32_to_bf16(const float* __restrict__ in,
                                                     __bf16* __restrict__ out, int n) {
    int gid = blockIdx.x * 256 + threadIdx.x;
    if (gid < n) out[gid] = (__bf16)in[gid];
}

__global__ __launch_bounds__(256) void k_zero_bf16(__bf16* __restrict__ p, int n) {
    int gid = blockIdx.x * 256 + threadIdx.x;
    if (gid < n) p[gid] = (__bf16)0.0f;
}

// ---------------------------------------------------------------------------
// TDM: DMA the [rows, 1024] bf16 row-major slab at gptr into LDS offset 0.
// D# per CDNA5 ISA (2-D tensor, groups 2/3 zero).
// This toolchain exposes the 6-arg builtin:
//   (u32x4 g0, i32x8 g1, i32x4 g2, i32x4 g3, i32x8 g4, i32 cpol)
// Issue from one wave; caller waits with s_wait_tensorcnt + barrier.
// ---------------------------------------------------------------------------
__device__ __forceinline__ void tdm_load_a(const __bf16* gptr, int rows) {
    unsigned long long ga = (unsigned long long)(size_t)gptr;
    u32x4 g0;
    g0[0] = 1u;                                        // count=1, user mode
    g0[1] = 0u;                                        // lds_addr = 0
    g0[2] = (unsigned)ga;                              // global_addr[31:0]
    g0[3] = (unsigned)((ga >> 32) & 0x1FFFFFFu)        // global_addr[56:32]
            | (2u << 30);                              // type = 2 ("image")
    i32x8 g1;
    g1[0] = (1 << 16);                                 // data_size = 1 -> 2 bytes
    g1[1] = (RNN_H & 0xFFFF) << 16;                    // tensor_dim0[15:0]
    g1[2] = (rows << 16) | (RNN_H >> 16);              // dim0[31:16] | dim1[15:0]
    g1[3] = (RNN_H << 16) | (rows >> 16);              // tile_dim0 | dim1[31:16]
    g1[4] = rows;                                      // tile_dim1 (tile_dim2=0)
    g1[5] = RNN_H;                                     // tensor_dim0_stride[31:0]
    g1[6] = 0;                                         // stride hi / dim1_stride lo
    g1[7] = 0;
    i32x4 g2 = {0, 0, 0, 0};
    i32x4 g3 = {0, 0, 0, 0};
    i32x8 g4 = {0, 0, 0, 0, 0, 0, 0, 0};
    __builtin_amdgcn_tensor_load_to_lds(g0, g1, g2, g3, g4, 0);
}

__device__ __forceinline__ void tdm_fetch_and_sync(const __bf16* gptr) {
    if ((threadIdx.x >> 5) == 0) {
        tdm_load_a(gptr, RNN_B);
        __builtin_amdgcn_s_wait_tensorcnt((short)0);
    }
    __syncthreads();
}

// ---------------------------------------------------------------------------
// WMMA helpers. A fragment (16-bit A 16x32 ISA layout): two contiguous 16B
// loads per lane; lane<16 takes K{0..7,16..23}, lane>=16 takes K{8..15,24..31}.
// ---------------------------------------------------------------------------
__device__ __forceinline__ v16bf make_afrag(const __bf16* p) {
    v8bf lo = *(const v8bf*)p;
    v8bf hi = *(const v8bf*)(p + 16);
    return __builtin_shufflevector(lo, hi,
                                   0, 1, 2, 3, 4, 5, 6, 7,
                                   8, 9, 10, 11, 12, 13, 14, 15);
}

__device__ __forceinline__ v8f wmma_bf16(v16bf a, v16bf b, v8f c) {
    return __builtin_amdgcn_wmma_f32_16x16x32_bf16(false, a, false, b,
                                                   (short)0, c, false, false);
}

// 2x2 tile GEMM: A in LDS ([64,1024] bf16 row-major), B packed in global.
//   a : wave M-pair index (0..1)  -> rows 32a..32a+31 (tiles 2a, 2a+1)
//   ntg0 : first of two adjacent global N tiles
__device__ __forceinline__ void gemm2x2(const __bf16* lds_a,
                                        const __bf16* __restrict__ Wp,
                                        int a, int ntg0, int lane,
                                        v8f& c00, v8f& c01, v8f& c10, v8f& c11) {
    const int kcN = RNN_H >> 5;  // 32
    int aoff = (lane >> 4) ? 8 : 0;
    const __bf16* ar0 = lds_a + (size_t)(a * 32 + (lane & 15)) * RNN_H + aoff;
    const __bf16* ar1 = ar0 + 16 * RNN_H;
    const __bf16* bp0 = Wp + (size_t)ntg0 * kcN * 512 + lane * 16;
    const __bf16* bp1 = bp0 + (size_t)kcN * 512;
    for (int kc = 0; kc < kcN; ++kc) {
        v16bf a0 = make_afrag(ar0);
        v16bf a1 = make_afrag(ar1);
        ar0 += 32; ar1 += 32;
        v16bf b0 = *(const v16bf*)bp0;
        v16bf b1 = *(const v16bf*)bp1;
        __builtin_prefetch((const void*)(bp0 + 512 * 4), 0, 3);
        __builtin_prefetch((const void*)(bp1 + 512 * 4), 0, 3);
        bp0 += 512; bp1 += 512;
        c00 = wmma_bf16(a0, b0, c00);
        c01 = wmma_bf16(a0, b1, c01);
        c10 = wmma_bf16(a1, b0, c10);
        c11 = wmma_bf16(a1, b1, c11);
    }
}

// C/D layout (16x16 f32, 8 VGPRs): lanes 0-15 -> M = v, N = lane;
//                                  lanes 16-31 -> M = v+8, N = lane-16.

// Layer 0: h0_out = tanh( h0_in @ Whh0 + W_ih[x[:,t]] + b_h[0] )
__global__ __launch_bounds__(256) void k_layer0(const __bf16* __restrict__ h0_in,
                                                const __bf16* __restrict__ Wp,
                                                const __bf16* __restrict__ Wih,
                                                const int* __restrict__ x,
                                                const float* __restrict__ bh0,
                                                __bf16* __restrict__ h0_out,
                                                int t) {
    extern __shared__ __bf16 lds_a[];
    int lane = threadIdx.x & 31;
    int wave = threadIdx.x >> 5;
    int a    = wave >> 2;
    int ntg0 = blockIdx.x * 8 + (wave & 3) * 2;

    tdm_fetch_and_sync(h0_in);

    v8f c00 = {}, c01 = {}, c10 = {}, c11 = {};
    gemm2x2(lds_a, Wp, a, ntg0, lane, c00, c01, c10, c11);

    int hi    = (lane >> 4) ? 8 : 0;
    int n0    = ntg0 * 16 + (lane & 15);
    int n1    = n0 + 16;
    float bi0 = bh0[n0];
    float bi1 = bh0[n1];
#pragma unroll
    for (int i = 0; i < 2; ++i) {
        const v8f& cx0 = i ? c10 : c00;
        const v8f& cx1 = i ? c11 : c01;
#pragma unroll
        for (int v = 0; v < 8; ++v) {
            int r  = a * 32 + i * 16 + hi + v;
            size_t gi = (size_t)x[r * RNN_S + t] * RNN_H;
            float g0 = (float)Wih[gi + n0];
            float g1 = (float)Wih[gi + n1];
            h0_out[(size_t)r * RNN_H + n0] = (__bf16)tanhf(cx0[v] + g0 + bi0);
            h0_out[(size_t)r * RNN_H + n1] = (__bf16)tanhf(cx1[v] + g1 + bi1);
        }
    }
}

// Layer 1: h1_out = tanh( h0_new @ W_h_h + h1_in @ Whh1 + b_h[1] )
// Two phases reuse the same 128KB LDS A buffer.
__global__ __launch_bounds__(256) void k_layer1(const __bf16* __restrict__ h0_new,
                                                const __bf16* __restrict__ Wpx,
                                                const __bf16* __restrict__ h1_in,
                                                const __bf16* __restrict__ Wp1,
                                                const float* __restrict__ bh1,
                                                __bf16* __restrict__ h1_out) {
    extern __shared__ __bf16 lds_a[];
    int lane = threadIdx.x & 31;
    int wave = threadIdx.x >> 5;
    int a    = wave >> 2;
    int ntg0 = blockIdx.x * 8 + (wave & 3) * 2;

    v8f c00 = {}, c01 = {}, c10 = {}, c11 = {};

    tdm_fetch_and_sync(h0_new);
    gemm2x2(lds_a, Wpx, a, ntg0, lane, c00, c01, c10, c11);
    __syncthreads();                       // all waves done reading LDS
    tdm_fetch_and_sync(h1_in);
    gemm2x2(lds_a, Wp1, a, ntg0, lane, c00, c01, c10, c11);

    int hi    = (lane >> 4) ? 8 : 0;
    int n0    = ntg0 * 16 + (lane & 15);
    int n1    = n0 + 16;
    float bi0 = bh1[n0];
    float bi1 = bh1[n1];
#pragma unroll
    for (int i = 0; i < 2; ++i) {
        const v8f& cx0 = i ? c10 : c00;
        const v8f& cx1 = i ? c11 : c01;
#pragma unroll
        for (int v = 0; v < 8; ++v) {
            size_t r = (size_t)(a * 32 + i * 16 + hi + v) * RNN_H;
            h1_out[r + n0] = (__bf16)tanhf(cx0[v] + bi0);
            h1_out[r + n1] = (__bf16)tanhf(cx1[v] + bi1);
        }
    }
}

// Output head: out_t = h1_new @ W_hy + b_y   (fp32, [64,256] per step)
__global__ __launch_bounds__(256) void k_out(const __bf16* __restrict__ h1,
                                             const __bf16* __restrict__ Wpy,
                                             const float* __restrict__ by,
                                             float* __restrict__ out) {
    extern __shared__ __bf16 lds_a[];
    int lane = threadIdx.x & 31;
    int wave = threadIdx.x >> 5;
    int a    = wave >> 2;
    int ntg0 = blockIdx.x * 8 + (wave & 3) * 2;   // 2 blocks -> ntg 0..15

    tdm_fetch_and_sync(h1);

    v8f c00 = {}, c01 = {}, c10 = {}, c11 = {};
    gemm2x2(lds_a, Wpy, a, ntg0, lane, c00, c01, c10, c11);

    int hi    = (lane >> 4) ? 8 : 0;
    int n0    = ntg0 * 16 + (lane & 15);
    int n1    = n0 + 16;
    float bi0 = by[n0];
    float bi1 = by[n1];
#pragma unroll
    for (int i = 0; i < 2; ++i) {
        const v8f& cx0 = i ? c10 : c00;
        const v8f& cx1 = i ? c11 : c01;
#pragma unroll
        for (int v = 0; v < 8; ++v) {
            size_t r = (size_t)(a * 32 + i * 16 + hi + v) * RNN_IN;
            out[r + n0] = cx0[v] + bi0;
            out[r + n1] = cx1[v] + bi1;
        }
    }
}

// ---------------------------------------------------------------------------
// Host launcher
// ---------------------------------------------------------------------------
extern "C" void kernel_launch(void* const* d_in, const int* in_sizes, int n_in,
                              void* d_out, int out_size, void* d_ws, size_t ws_size,
                              hipStream_t stream) {
    (void)in_sizes; (void)n_in; (void)out_size; (void)ws_size;

    const int*   x     = (const int*)d_in[0];    // [64, 512]
    const float* W_ih  = (const float*)d_in[1];  // [256, 1024]
    const float* W_hh  = (const float*)d_in[2];  // [2, 1024, 1024]
    const float* W_h_h = (const float*)d_in[3];  // [1, 1024, 1024]
    const float* b_h   = (const float*)d_in[4];  // [2, 1024]
    const float* W_hy  = (const float*)d_in[5];  // [1024, 256]
    const float* b_y   = (const float*)d_in[6];  // [256]
    float*       out   = (float*)d_out;          // [512, 64, 256]

    __bf16* ws = (__bf16*)d_ws;
    const size_t HH = (size_t)RNN_H * RNN_H;
    const size_t HY = (size_t)RNN_H * RNN_IN;
    const size_t IH = (size_t)RNN_IN * RNN_H;
    const size_t HB = (size_t)RNN_B * RNN_H;
    __bf16* whh0_p = ws;
    __bf16* whh1_p = whh0_p + HH;
    __bf16* whx_p  = whh1_p + HH;
    __bf16* why_p  = whx_p + HH;
    __bf16* wih_bf = why_p + HY;
    __bf16* h0a    = wih_bf + IH;
    __bf16* h1a    = h0a + HB;
    __bf16* h0b    = h1a + HB;
    __bf16* h1b    = h0b + HB;

    {
        int gHH = (int)((HH + 255) / 256);
        int gHY = (int)((HY + 255) / 256);
        int gIH = (int)((IH + 255) / 256);
        k_pack_b<<<gHH, 256, 0, stream>>>(W_hh,      whh0_p, RNN_H, RNN_H);
        k_pack_b<<<gHH, 256, 0, stream>>>(W_hh + HH, whh1_p, RNN_H, RNN_H);
        k_pack_b<<<gHH, 256, 0, stream>>>(W_h_h,     whx_p,  RNN_H, RNN_H);
        k_pack_b<<<gHY, 256, 0, stream>>>(W_hy,      why_p,  RNN_H, RNN_IN);
        k_f32_to_bf16<<<gIH, 256, 0, stream>>>(W_ih, wih_bf, (int)IH);
        k_zero_bf16<<<(int)((2 * HB + 255) / 256), 256, 0, stream>>>(h0a, (int)(2 * HB));
    }

    const __bf16* h0_in = h0a;
    const __bf16* h1_in = h1a;
    __bf16*       h0_out = h0b;
    __bf16*       h1_out = h1b;

    for (int t = 0; t < RNN_S; ++t) {
        k_layer0<<<8, 256, LDS_A_BYTES, stream>>>(h0_in, whh0_p, wih_bf, x,
                                                  b_h, h0_out, t);
        k_layer1<<<8, 256, LDS_A_BYTES, stream>>>(h0_out, whx_p, h1_in, whh1_p,
                                                  b_h + RNN_H, h1_out);
        k_out<<<2, 256, LDS_A_BYTES, stream>>>(h1_out, why_p, b_y,
                                               out + (size_t)t * RNN_B * RNN_IN);
        const __bf16* t0 = h0_in; h0_in = h0_out; h0_out = (__bf16*)t0;
        const __bf16* t1 = h1_in; h1_in = h1_out; h1_out = (__bf16*)t1;
    }
}
